// AdditiveAttention_5068061409831
// MI455X (gfx1250) — matrix operand order repaired and checked by reading, not compiled, not run on hardware
//
#include <hip/hip_runtime.h>
#include <hip/hip_bf16.h>

typedef __attribute__((ext_vector_type(16))) _Float16 v16h;
typedef __attribute__((ext_vector_type(8)))  _Float16 v8h;
typedef __attribute__((ext_vector_type(8)))  float    v8f;

#define NEGF (-1e6f)

// Hardware tanh: CDNA5 has V_TANH_F32 in the transcendental unit (same class
// as V_EXP_F32). Prefer the clang builtin; fall back to inline asm with a
// trailing v_nop to satisfy the "1 independent op after TRANS" hazard rule.
__device__ __forceinline__ float fast_tanh(float x) {
#if __has_builtin(__builtin_amdgcn_tanhf)
    return __builtin_amdgcn_tanhf(x);
#else
    float r;
    asm volatile("v_tanh_f32 %0, %1\n\tv_nop" : "=v"(r) : "v"(x));
    return r;
#endif
}

// ---------------------------------------------------------------------------
// Elementwise f32 -> f16 conversion
// ---------------------------------------------------------------------------
__global__ void aa_cvt_f16_kernel(const float* __restrict__ in,
                                  _Float16* __restrict__ out, int n) {
    int i = blockIdx.x * blockDim.x + threadIdx.x;
    if (i < n) out[i] = (_Float16)in[i];
}

// ---------------------------------------------------------------------------
// Transpose + convert: in[b][r][c] (f32, R x C row-major) -> out[b][c][r] (f16)
// ---------------------------------------------------------------------------
__global__ void aa_transpose_cvt_kernel(const float* __restrict__ in,
                                        _Float16* __restrict__ out,
                                        int R, int Ccols, long ibs, long obs) {
    int i = blockIdx.x * blockDim.x + threadIdx.x;
    int total = R * Ccols;
    if (i >= total) return;
    int r = i / Ccols;
    int c = i - r * Ccols;
    out[(size_t)blockIdx.y * obs + (size_t)c * R + r] =
        (_Float16)in[(size_t)blockIdx.y * ibs + i];
}

// ---------------------------------------------------------------------------
// Batched GEMM via V_WMMA_F32_16X16X32_F16.
//   C[M x N] (f32, row-major, ld = N) = A[M x Kd] (f16, row-major, ld = Kd)
//                                     * B[Kd x N] where B is stored TRANSPOSED:
//                                       BT[N x Kd] (f16, row-major, ld = Kd)
// One wave per 16x16 C tile; 4 waves (128 threads) per block.
// Fragment construction follows CDNA5 ISA 7.12.2:
//   A 16x32 f16 : lane(0-15)=row M, half=lane>>4 selects K sub-blocks
//                 a[e] = A[m][kk + (e<8 ? e : e+8) + 8*half]   -> two b128 loads
//   B 32x16 f16 : n = lane&15, b[e] = BT[n][kk + (e<8 ? e : e+8) + 8*half] -> two b128 loads
//   C 16x16 f32 : vgpr r -> row = 16*tm + r + 8*half, col = 16*tn + (lane&15)
// ---------------------------------------------------------------------------
__global__ void aa_gemm_f16_wmma_kernel(const _Float16* __restrict__ A,
                                        const _Float16* __restrict__ BT,
                                        float* __restrict__ C,
                                        int M, int N, int Kd,
                                        long sA, long sB, long sC) {
    A  += (size_t)blockIdx.y * sA;
    BT += (size_t)blockIdx.y * sB;
    C  += (size_t)blockIdx.y * sC;

    const int wave = threadIdx.x >> 5;
    const int lane = threadIdx.x & 31;
    const int half = lane >> 4;       // 0 or 1
    const int l16  = lane & 15;

    const int tilesM = M >> 4;
    const int tilesN = N >> 4;
    const int t = blockIdx.x * 4 + wave;          // wave-uniform
    if (t >= tilesM * tilesN) return;             // wave-uniform exit
    const int tm = t % tilesM;
    const int tn = t / tilesM;

    const int c0 = half * 8;
    const _Float16* arowBase = A  + (size_t)(tm * 16 + l16) * Kd;
    const _Float16* browBase = BT + (size_t)(tn * 16 + l16) * Kd + half * 8;

    v8f acc = {};
    for (int kk = 0; kk < Kd; kk += 32) {
        v8h alo = *(const v8h*)(arowBase + kk + c0);
        v8h ahi = *(const v8h*)(arowBase + kk + 16 + c0);
        v8h b0  = *(const v8h*)(browBase + kk);
        v8h b1  = *(const v8h*)(browBase + kk + 16);
        v16h af, bf;
#pragma unroll
        for (int i = 0; i < 8; ++i) {
            af[i]     = alo[i];
            af[i + 8] = ahi[i];
            bf[i]     = b0[i];
            bf[i + 8] = b1[i];
        }
        acc = __builtin_amdgcn_wmma_f32_16x16x32_f16(
            /*neg_a=*/false, af, /*neg_b=*/false, bf,
            /*c_mod=*/(short)0, acc, /*reuse_a=*/false, /*reuse_b=*/false);
    }

    float* cbase = C + (size_t)(tm * 16 + half * 8) * N + tn * 16 + l16;
#pragma unroll
    for (int r = 0; r < 8; ++r) cbase[(size_t)r * N] = acc[r];
}

// ---------------------------------------------------------------------------
// Fused scores + masked softmax:
//   scores[b,q,k] = sum_h tanh(qf[b,q,h] + kf[b,k,h]) * wv[h]
//   masked softmax over k (k >= valid_len[b] -> -1e6), output f16 attn.
// One 256-thread block per (b,q); each thread owns 4 k-positions.
// B=8, Q=256, K=1024, H=128 hardcoded.
// ---------------------------------------------------------------------------
__global__ void aa_scores_softmax_kernel(const float* __restrict__ qf,
                                         const float* __restrict__ kf,
                                         const float* __restrict__ wv,
                                         const int* __restrict__ vlen,
                                         _Float16* __restrict__ attn) {
    const int K = 1024, H = 128;
    const int bq  = blockIdx.x;      // b*256 + q
    const int b   = bq >> 8;
    const int tid = threadIdx.x;

    __shared__ __align__(16) float sq[128];
    __shared__ __align__(16) float sw[128];
    __shared__ float red[256];

    if (tid < H) {
        sq[tid] = qf[(size_t)bq * H + tid];
        sw[tid] = wv[tid];
    }
    __syncthreads();

    const int vl = vlen[b];
    float s[4];
#pragma unroll
    for (int j = 0; j < 4; ++j) {
        const int k_ = tid + j * 256;
        const float4* krow = (const float4*)(kf + ((size_t)b * K + k_) * H);
        const float4* q4 = (const float4*)sq;
        const float4* w4 = (const float4*)sw;
        float acc = 0.f;
#pragma unroll 4
        for (int h4 = 0; h4 < H / 4; ++h4) {
            float4 kv = krow[h4];
            float4 qv = q4[h4];
            float4 wvv = w4[h4];
            acc += fast_tanh(qv.x + kv.x) * wvv.x;
            acc += fast_tanh(qv.y + kv.y) * wvv.y;
            acc += fast_tanh(qv.z + kv.z) * wvv.z;
            acc += fast_tanh(qv.w + kv.w) * wvv.w;
        }
        s[j] = (k_ < vl) ? acc : NEGF;
    }

    // block max reduction
    float lmax = fmaxf(fmaxf(s[0], s[1]), fmaxf(s[2], s[3]));
    red[tid] = lmax;
    __syncthreads();
    for (int st = 128; st > 0; st >>= 1) {
        if (tid < st) red[tid] = fmaxf(red[tid], red[tid + st]);
        __syncthreads();
    }
    const float mx = red[0];
    __syncthreads();

    float e[4];
    float lsum = 0.f;
#pragma unroll
    for (int j = 0; j < 4; ++j) {
        e[j] = __expf(s[j] - mx);
        lsum += e[j];
    }
    red[tid] = lsum;
    __syncthreads();
    for (int st = 128; st > 0; st >>= 1) {
        if (tid < st) red[tid] = red[tid] + red[tid + st];
        __syncthreads();
    }
    const float inv = 1.f / red[0];

#pragma unroll
    for (int j = 0; j < 4; ++j) {
        attn[(size_t)bq * K + tid + j * 256] = (_Float16)(e[j] * inv);
    }
}

// ---------------------------------------------------------------------------
// Host-side orchestration
// ---------------------------------------------------------------------------
extern "C" void kernel_launch(void* const* d_in, const int* in_sizes, int n_in,
                              void* d_out, int out_size, void* d_ws, size_t ws_size,
                              hipStream_t stream) {
    const int B = 8, Q = 256, K = 1024, Dq = 256, Dk = 256, Dv = 128, H = 128;

    const float* queries = (const float*)d_in[0];  // [B,Q,Dq]
    const float* keys    = (const float*)d_in[1];  // [B,K,Dk]
    const float* values  = (const float*)d_in[2];  // [B,K,Dv]
    const int*   vlen    = (const int*)  d_in[3];  // [B]
    const float* W_q     = (const float*)d_in[4];  // [Dq,H]
    const float* W_k     = (const float*)d_in[5];  // [Dk,H]
    const float* w_v     = (const float*)d_in[6];  // [H]
    float* out = (float*)d_out;                    // [B,Q,Dv]

    // workspace bump allocator (256B aligned)
    char* ws = (char*)d_ws;
    size_t off = 0;
    auto alloc = [&](size_t bytes) -> void* {
        void* p = ws + off;
        off = (off + bytes + 255) & ~(size_t)255;
        return p;
    };
    _Float16* q16   = (_Float16*)alloc((size_t)B * Q * Dq * 2);   // 1 MB
    _Float16* k16   = (_Float16*)alloc((size_t)B * K * Dk * 2);   // 4 MB
    _Float16* WqT16 = (_Float16*)alloc((size_t)H * Dq * 2);       // 64 KB
    _Float16* WkT16 = (_Float16*)alloc((size_t)H * Dk * 2);       // 64 KB
    float*    qf    = (float*)   alloc((size_t)B * Q * H * 4);    // 1 MB
    float*    kf    = (float*)   alloc((size_t)B * K * H * 4);    // 4 MB
    _Float16* att16 = (_Float16*)alloc((size_t)B * Q * K * 2);    // 4 MB
    _Float16* VT16  = (_Float16*)alloc((size_t)B * Dv * K * 2);   // 2 MB

    // 1) convert queries/keys to f16
    {
        int n1 = B * Q * Dq;
        aa_cvt_f16_kernel<<<(n1 + 255) / 256, 256, 0, stream>>>(queries, q16, n1);
        int n2 = B * K * Dk;
        aa_cvt_f16_kernel<<<(n2 + 255) / 256, 256, 0, stream>>>(keys, k16, n2);
    }
    // 2) transpose+convert weight matrices and values
    {
        int nW = Dq * H;  // 32768
        aa_transpose_cvt_kernel<<<dim3((nW + 255) / 256, 1), 256, 0, stream>>>(
            W_q, WqT16, Dq, H, 0, 0);
        aa_transpose_cvt_kernel<<<dim3((nW + 255) / 256, 1), 256, 0, stream>>>(
            W_k, WkT16, Dk, H, 0, 0);
        int nV = K * Dv;  // 131072 per batch
        aa_transpose_cvt_kernel<<<dim3((nV + 255) / 256, B), 256, 0, stream>>>(
            values, VT16, K, Dv, (long)nV, (long)nV);
    }
    // 3) q projection: qf[B*Q, H] = q16[B*Q, Dq] * WqT16^T   (batch folded)
    {
        int M = B * Q, N = H, Kd = Dq;
        int tiles = (M >> 4) * (N >> 4);
        aa_gemm_f16_wmma_kernel<<<dim3((tiles + 3) / 4, 1), 128, 0, stream>>>(
            q16, WqT16, qf, M, N, Kd, 0, 0, 0);
    }
    // 4) k projection: kf[B*K, H] = k16[B*K, Dk] * WkT16^T
    {
        int M = B * K, N = H, Kd = Dk;
        int tiles = (M >> 4) * (N >> 4);
        aa_gemm_f16_wmma_kernel<<<dim3((tiles + 3) / 4, 1), 128, 0, stream>>>(
            k16, WkT16, kf, M, N, Kd, 0, 0, 0);
    }
    // 5) fused scores (hw tanh) + masked softmax -> f16 attention
    aa_scores_softmax_kernel<<<B * Q, 256, 0, stream>>>(qf, kf, w_v, vlen, att16);

    // 6) out[b] = attn[b] (256x1024) * values[b] (1024x128), batched
    {
        int M = Q, N = Dv, Kd = K;
        int tiles = (M >> 4) * (N >> 4);  // 128
        aa_gemm_f16_wmma_kernel<<<dim3((tiles + 3) / 4, B), 128, 0, stream>>>(
            att16, VT16, out, M, N, Kd,
            (long)Q * K, (long)Dv * K, (long)Q * Dv);
    }
}
